// FactoredGRU_58591943852346
// MI455X (gfx1250) — matrix operand order repaired and checked by reading, not compiled, not run on hardware
//
#include <hip/hip_runtime.h>
#include <cmath>

// ---------------------------------------------------------------------------
// Types for CDNA5 WMMA (wave32)
// ---------------------------------------------------------------------------
typedef __bf16 bf16;
typedef bf16  bf16x16 __attribute__((ext_vector_type(16)));
typedef float f32x8   __attribute__((ext_vector_type(8)));
typedef unsigned int u32x4 __attribute__((ext_vector_type(4)));

// Exact pointee type of the async-load builtin params (clang prints it as
// 'int __attribute__((vector_size(16)))').
typedef int v4i_gcc __attribute__((vector_size(16)));
typedef __attribute__((address_space(1))) v4i_gcc* g_v4i_ptr;  // __device__
typedef __attribute__((address_space(3))) v4i_gcc* l_v4i_ptr;  // __shared__

union Frag16 { bf16x16 v; u32x4 q[2]; };
union Acc8   { f32x8 v;  float f[8];  };

#define DEVFN __device__ __forceinline__

// ---------------------------------------------------------------------------
// Async global->LDS (CDNA5 GLOBAL_LOAD_ASYNC_TO_LDS_B128, ASYNCcnt-tracked).
// Guarded: fall back to sync load + ds_store_b128 if the builtin is absent
// (also keeps the host pass free of amdgcn builtins).
// ---------------------------------------------------------------------------
#if defined(__gfx1250__) && __has_builtin(__builtin_amdgcn_global_load_async_to_lds_b128)
#define ASYNC_G2L 1
#else
#define ASYNC_G2L 0
#endif

DEVFN void copy16_g2l(const bf16* __restrict__ g, bf16* __restrict__ l) {
#if ASYNC_G2L
  // Generic LDS pointers carry the LDS byte offset in the low 32 bits;
  // generic global pointers equal the 64-bit VA.
  __builtin_amdgcn_global_load_async_to_lds_b128(
      (g_v4i_ptr)(uintptr_t)g, (l_v4i_ptr)(uintptr_t)l, 0, 0);
#else
  *(u32x4*)l = *(const u32x4*)g;  // global_load_b128 + ds_store_b128
#endif
}

template <int N>
DEVFN void async_wait() {
#if ASYNC_G2L
#if __has_builtin(__builtin_amdgcn_s_wait_asynccnt)
  __builtin_amdgcn_s_wait_asynccnt((short)N);
#else
  asm volatile("s_wait_asynccnt %0" ::"i"(N) : "memory");
#endif
#endif
}

// ---------------------------------------------------------------------------
// WMMA fragment layouts used here (wave32; contiguous k order, same for A and B, not the ISA table order):
//   A (16x32 bf16, row-major src):   lane l: m=l&15, hl=l>>4
//     a[i] = A[m][k0 + hl*16 + i]   (contiguous 32 bytes)
//   B (32x16 bf16) from TRANSPOSED weights Bt[N][K]: lane l: n=l&15, hl=l>>4
//     b[j] = Bt[n][k0 + hl*16 + j]   (contiguous 32 bytes)
//   C/D (16x16 f32): lane l: n=l&15, hl=l>>4, c[v] = C[hl*8+v][n]
// ---------------------------------------------------------------------------
DEVFN Frag16 load_a_frag(const bf16* __restrict__ A, size_t row, int k0, int lda, int lane) {
  const int hl = lane >> 4;
  const bf16* p = A + row * (size_t)lda + k0 + hl * 16;
  Frag16 f;
  f.q[0] = *(const u32x4*)(p);
  f.q[1] = *(const u32x4*)(p + 8);
  return f;
}

DEVFN Frag16 load_b_frag(const bf16* __restrict__ Bt, size_t col, int k0, int ldb, int lane) {
  const int hl = lane >> 4;
  const bf16* p = Bt + col * (size_t)ldb + k0 + hl * 16;
  Frag16 f;
  f.q[0] = *(const u32x4*)(p);
  f.q[1] = *(const u32x4*)(p + 8);
  return f;
}

DEVFN f32x8 wmma_bf16(bf16x16 a, bf16x16 b, f32x8 c) {
  // (neg_a, A, neg_b, B, c_mod, C, reuse_a, reuse_b)
  return __builtin_amdgcn_wmma_f32_16x16x32_bf16(false, a, false, b, (short)0, c, false, false);
}

DEVFN float sigmoidf_(float x) { return 1.0f / (1.0f + expf(-x)); }

// ---------------------------------------------------------------------------
// Weight convert + transpose: W[K][N] fp32 -> Wt[N][K] bf16
// ---------------------------------------------------------------------------
__global__ void convert_transpose_kernel(const float* __restrict__ W,
                                         bf16* __restrict__ Wt, int K, int N) {
  size_t idx = (size_t)blockIdx.x * blockDim.x + threadIdx.x;
  if (idx < (size_t)K * N) {
    int k = (int)(idx / N);
    int n = (int)(idx % N);
    Wt[(size_t)n * K + k] = (bf16)W[idx];
  }
}

// ---------------------------------------------------------------------------
// Build step inputs x[b*T+t][:] in bf16:
//   t==0 -> features[b][:], t>0 -> emb_table[captions[b][t-1]][:]
// ---------------------------------------------------------------------------
__global__ void build_x_kernel(const int* __restrict__ captions,
                               const float* __restrict__ features,
                               const float* __restrict__ emb_table,
                               bf16* __restrict__ x_bf) {
  const int m = blockIdx.x;  // 0..4095, m = b*64 + t
  const int b = m >> 6, t = m & 63;
  const float* src = (t == 0) ? (features + (size_t)b * 512)
                              : (emb_table + (size_t)captions[b * 64 + (t - 1)] * 512);
  for (int e = threadIdx.x; e < 512; e += blockDim.x)
    x_bf[(size_t)m * 512 + e] = (bf16)src[e];
}

__global__ void init_h_kernel(const float* __restrict__ h0,
                              float* __restrict__ h, bf16* __restrict__ h_bf) {
  int i = blockIdx.x * blockDim.x + threadIdx.x;
  if (i < 64 * 512) {
    float v = h0[i];
    h[i] = v;
    h_bf[i] = (bf16)v;
  }
}

// ---------------------------------------------------------------------------
// Tiled bf16 WMMA GEMM with double-buffered async LDS staging:
//   C[M,N] = A[M,K] (bf16, lda) @ Bt[N,K]^T + bias[n] + row_add[m/row_div][n]
// Block 256 thr = 8 waves; tile 128x128; wave tile 64x32 (4x2 C-tiles);
// K staged in 32-deep slabs through LDS (padded stride 40 => conflict-free
// ds_load_b128 fragment reads), next slab prefetched via ASYNCcnt while the
// current slab feeds 64 WMMAs.
// OUT_BF16: store bf16 (chained GEMMs) else fp32.
// REMAP:    m = t*64+b -> output row b*64+t (logits einsum tbh,hv->btv).
// ---------------------------------------------------------------------------
template <bool OUT_BF16, bool REMAP>
__global__ __launch_bounds__(256)
void gemm_bf16_kernel(const bf16* __restrict__ A, int lda,
                      const bf16* __restrict__ Bt,
                      const float* __restrict__ bias,
                      const float* __restrict__ row_add, int row_div,
                      float* __restrict__ Cf, bf16* __restrict__ Cb,
                      int M, int N, int K) {
  constexpr int KT = 32;   // K depth per stage
  constexpr int LDT = 40;  // padded LDS row stride (elements) -> 80B = 20 banks
  __shared__ bf16 sA[2][128 * LDT];
  __shared__ bf16 sB[2][128 * LDT];

  const int tid = threadIdx.x;
  const int lane = tid & 31;
  const int wave = tid >> 5;
  const int wm = wave & 1;   // 2 waves along M
  const int wn = wave >> 1;  // 4 waves along N
  const int block_row = blockIdx.y * 128;
  const int block_col = blockIdx.x * 128;
  const int hl = lane >> 4;
  const int l16 = lane & 15;

  // Cooperative stage: 512 16B chunks per matrix; 2 per thread per matrix.
  auto stage = [&](int buf, int k0) {
#pragma unroll
    for (int i = 0; i < 2; ++i) {
      const int c = tid + i * 256;         // 0..511
      const int row = c >> 2;              // 0..127
      const int q = (c & 3) * 8;           // element offset within slab (16B units)
      int ra = block_row + row;
      if (ra >= M) ra = M - 1;             // safe clamp; stores are guarded
      copy16_g2l(A + (size_t)ra * lda + k0 + q, &sA[buf][row * LDT + q]);
      copy16_g2l(Bt + (size_t)(block_col + row) * K + k0 + q, &sB[buf][row * LDT + q]);
    }
  };

  Acc8 acc[4][2];
#pragma unroll
  for (int mt = 0; mt < 4; ++mt)
#pragma unroll
    for (int nt = 0; nt < 2; ++nt)
#pragma unroll
      for (int i = 0; i < 8; ++i) acc[mt][nt].f[i] = 0.0f;

  const int nstage = K / KT;
  stage(0, 0);
  for (int kt = 0; kt < nstage; ++kt) {
    const int buf = kt & 1;
    if (kt + 1 < nstage) {
      stage(buf ^ 1, (kt + 1) * KT);  // prefetch next slab (4 async ops/wave)
      async_wait<4>();                // current slab's 4 ops done; next in flight
    } else {
      async_wait<0>();
    }
    __syncthreads();

    Frag16 bfrag[2];
#pragma unroll
    for (int nt = 0; nt < 2; ++nt) {
      const bf16* p = &sB[buf][(wn * 32 + nt * 16 + l16) * LDT + hl * 16];
      bfrag[nt].q[0] = *(const u32x4*)p;
      bfrag[nt].q[1] = *(const u32x4*)(p + 8);
    }
#pragma unroll
    for (int mt = 0; mt < 4; ++mt) {
      const bf16* p = &sA[buf][(wm * 64 + mt * 16 + l16) * LDT + hl * 16];
      Frag16 af;
      af.q[0] = *(const u32x4*)p;
      af.q[1] = *(const u32x4*)(p + 8);
#pragma unroll
      for (int nt = 0; nt < 2; ++nt)
        acc[mt][nt].v = wmma_bf16(af.v, bfrag[nt].v, acc[mt][nt].v);
    }
    __syncthreads();
  }

#pragma unroll
  for (int mt = 0; mt < 4; ++mt)
#pragma unroll
    for (int nt = 0; nt < 2; ++nt)
#pragma unroll
      for (int v = 0; v < 8; ++v) {
        const int r = block_row + wm * 64 + mt * 16 + hl * 8 + v;
        const int c = block_col + wn * 32 + nt * 16 + l16;
        if (r < M) {
          float val = acc[mt][nt].f[v];
          if (bias) val += bias[c];
          if (row_add) val += row_add[(size_t)(r / row_div) * N + c];
          size_t orow = REMAP ? (size_t)((r & 63) * 64 + (r >> 6)) : (size_t)r;
          if (OUT_BF16) Cb[orow * (size_t)N + c] = (bf16)val;
          else          Cf[orow * (size_t)N + c] = val;
        }
      }
}

// ---------------------------------------------------------------------------
// GRU recurrence, step t, phase 1: z,r gates (register double-buffered).
//   z = sigmoid(a_z[b*T+t] + h@Wz + bz); r = sigmoid(a_r[...] + h@Wr + br)
//   store z (fp32) and rh = bf16(r*h).
// Grid: 16 blocks x 256 thr; block = 64 rows x 32 cols; wave tile 16x16.
// Fragments for K-step k+1 are loaded (ping-pong regs) before the WMMAs of
// step k, so the steady state overlaps global loads with matrix math instead
// of stalling on s_wait_loadcnt 0.
// ---------------------------------------------------------------------------
__global__ __launch_bounds__(256)
void gru_phase1_kernel(int t,
                       const float* __restrict__ h, const bf16* __restrict__ h_bf,
                       const bf16* __restrict__ Wz_t, const bf16* __restrict__ Wr_t,
                       const float* __restrict__ Wz_b, const float* __restrict__ Wr_b,
                       const float* __restrict__ a_z, const float* __restrict__ a_r,
                       float* __restrict__ z_ws, bf16* __restrict__ rh_bf) {
  const int lane = threadIdx.x & 31;
  const int wave = threadIdx.x >> 5;
  const int row0 = (wave & 3) * 16;                     // 4 waves over M=64
  const int col0 = blockIdx.x * 32 + (wave >> 2) * 16;  // 16 blk x 2 waves, N=512
  const int hl = lane >> 4;
  const int l16 = lane & 15;

  Acc8 accz, accr;
#pragma unroll
  for (int i = 0; i < 8; ++i) { accz.f[i] = 0.0f; accr.f[i] = 0.0f; }

  Frag16 aF[2], bzF[2], brF[2];
  aF[0]  = load_a_frag(h_bf, (size_t)(row0 + l16), 0, 512, lane);
  bzF[0] = load_b_frag(Wz_t, (size_t)(col0 + l16), 0, 512, lane);
  brF[0] = load_b_frag(Wr_t, (size_t)(col0 + l16), 0, 512, lane);

#pragma unroll
  for (int ks = 0; ks < 16; ++ks) {
    const int cur = ks & 1, nxt = cur ^ 1;
    if (ks + 1 < 16) {
      const int k1 = (ks + 1) * 32;
      aF[nxt]  = load_a_frag(h_bf, (size_t)(row0 + l16), k1, 512, lane);
      bzF[nxt] = load_b_frag(Wz_t, (size_t)(col0 + l16), k1, 512, lane);
      brF[nxt] = load_b_frag(Wr_t, (size_t)(col0 + l16), k1, 512, lane);
    }
    accz.v = wmma_bf16(aF[cur].v, bzF[cur].v, accz.v);
    accr.v = wmma_bf16(aF[cur].v, brF[cur].v, accr.v);
  }

#pragma unroll
  for (int v = 0; v < 8; ++v) {
    const int b = row0 + hl * 8 + v;  // batch row (0..63)
    const int c = col0 + l16;         // hidden column
    const size_t am = ((size_t)b * 64 + t) * 512 + c;
    const size_t hm = (size_t)b * 512 + c;
    float zz = sigmoidf_(accz.f[v] + a_z[am] + Wz_b[c]);
    float rr = sigmoidf_(accr.f[v] + a_r[am] + Wr_b[c]);
    z_ws[hm] = zz;
    rh_bf[hm] = (bf16)(rr * h[hm]);
  }
}

// ---------------------------------------------------------------------------
// GRU step t, phase 2: n gate + h update (register double-buffered).
//   n = tanh(a_n[b*T+t] + (r*h)@Wn + bn); h = (1-z)*h + z*n
//   write h (fp32), h_bf, and hs_bf[t*64+b].
// ---------------------------------------------------------------------------
__global__ __launch_bounds__(256)
void gru_phase2_kernel(int t,
                       float* __restrict__ h, bf16* __restrict__ h_bf,
                       const bf16* __restrict__ Wn_t, const float* __restrict__ Wn_b,
                       const float* __restrict__ a_n,
                       const float* __restrict__ z_ws, const bf16* __restrict__ rh_bf,
                       bf16* __restrict__ hs_bf) {
  const int lane = threadIdx.x & 31;
  const int wave = threadIdx.x >> 5;
  const int row0 = (wave & 3) * 16;
  const int col0 = blockIdx.x * 32 + (wave >> 2) * 16;
  const int hl = lane >> 4;
  const int l16 = lane & 15;

  Acc8 accn;
#pragma unroll
  for (int i = 0; i < 8; ++i) accn.f[i] = 0.0f;

  Frag16 aF[2], bnF[2];
  aF[0]  = load_a_frag(rh_bf, (size_t)(row0 + l16), 0, 512, lane);
  bnF[0] = load_b_frag(Wn_t, (size_t)(col0 + l16), 0, 512, lane);

#pragma unroll
  for (int ks = 0; ks < 16; ++ks) {
    const int cur = ks & 1, nxt = cur ^ 1;
    if (ks + 1 < 16) {
      const int k1 = (ks + 1) * 32;
      aF[nxt]  = load_a_frag(rh_bf, (size_t)(row0 + l16), k1, 512, lane);
      bnF[nxt] = load_b_frag(Wn_t, (size_t)(col0 + l16), k1, 512, lane);
    }
    accn.v = wmma_bf16(aF[cur].v, bnF[cur].v, accn.v);
  }

#pragma unroll
  for (int v = 0; v < 8; ++v) {
    const int b = row0 + hl * 8 + v;
    const int c = col0 + l16;
    const size_t am = ((size_t)b * 64 + t) * 512 + c;
    const size_t hm = (size_t)b * 512 + c;
    float nv = tanhf(accn.f[v] + a_n[am] + Wn_b[c]);
    float zz = z_ws[hm];
    float hn = (1.0f - zz) * h[hm] + zz * nv;
    h[hm] = hn;
    h_bf[hm] = (bf16)hn;
    hs_bf[((size_t)t * 64 + b) * 512 + c] = (bf16)hn;  // row m = t*B + b
  }
}

// ---------------------------------------------------------------------------
// Host launcher
// ---------------------------------------------------------------------------
extern "C" void kernel_launch(void* const* d_in, const int* in_sizes, int n_in,
                              void* d_out, int out_size, void* d_ws, size_t ws_size,
                              hipStream_t stream) {
  (void)in_sizes; (void)n_in; (void)out_size; (void)ws_size;
  constexpr int Vv = 16384, Ee = 512, Hh = 512, FDd = 512, Bb = 64, Tt = 64;
  constexpr int Mrows = Bb * Tt;  // 4096

  const int*   captions  = (const int*)d_in[0];
  const float* features  = (const float*)d_in[1];
  const float* h0        = (const float*)d_in[2];
  const float* emb_table = (const float*)d_in[3];
  const float* C_w       = (const float*)d_in[4];
  const float* C_b       = (const float*)d_in[5];
  auto gin = [&](int g, int off) { return (const float*)d_in[6 + g * 10 + off]; };
  // per-gate offsets: Vw 0, Vb 1, Sw 2, Sb 3, Fw 4, Fb 5, Uw 6, Ub 7, Ww 8, Wb 9

  char* p = (char*)d_ws;
  auto carve = [&](size_t bytes) -> void* {
    void* r = (void*)p;
    p += (bytes + 255) & ~(size_t)255;
    return r;
  };

  bf16* x_bf = (bf16*)carve((size_t)Mrows * Ee * sizeof(bf16));
  bf16* wt[15];  // [gate*5 + {V,S,F,U,W}] transposed bf16 512x512
  for (int i = 0; i < 15; ++i) wt[i] = (bf16*)carve((size_t)512 * 512 * sizeof(bf16));
  bf16* cw_t = (bf16*)carve((size_t)Vv * Hh * sizeof(bf16));
  bf16* g1 = (bf16*)carve((size_t)Mrows * FDd * sizeof(bf16));
  bf16* g2 = (bf16*)carve((size_t)Mrows * FDd * sizeof(bf16));
  float* fproj[3];
  for (int g = 0; g < 3; ++g) fproj[g] = (float*)carve((size_t)Bb * FDd * sizeof(float));
  float* a_g[3];
  for (int g = 0; g < 3; ++g) a_g[g] = (float*)carve((size_t)Mrows * Hh * sizeof(float));
  float* hbuf  = (float*)carve((size_t)Bb * Hh * sizeof(float));
  bf16*  h_bf  = (bf16*)carve((size_t)Bb * Hh * sizeof(bf16));
  float* z_ws  = (float*)carve((size_t)Bb * Hh * sizeof(float));
  bf16*  rh_bf = (bf16*)carve((size_t)Bb * Hh * sizeof(bf16));
  bf16*  hs_bf = (bf16*)carve((size_t)Mrows * Hh * sizeof(bf16));

  // 1) Convert+transpose all weights to bf16 [N,K]
  for (int g = 0; g < 3; ++g) {
    const float* mats[5] = { gin(g, 0), gin(g, 2), gin(g, 4), gin(g, 6), gin(g, 8) };
    for (int j = 0; j < 5; ++j)
      convert_transpose_kernel<<<(512 * 512 + 255) / 256, 256, 0, stream>>>(
          mats[j], wt[g * 5 + j], 512, 512);
  }
  convert_transpose_kernel<<<(int)(((size_t)Hh * Vv + 255) / 256), 256, 0, stream>>>(
      C_w, cw_t, Hh, Vv);

  // 2) Build step inputs in bf16
  build_x_kernel<<<Mrows, 128, 0, stream>>>(captions, features, emb_table, x_bf);

  // 3) Gate pre-activations:  a_g = ((x@V + Vb)@S + Sb + (f@F + Fb)) @ U + Ub
  for (int g = 0; g < 3; ++g) {
    // fproj = features @ F + Fb  (A rows are x_bf rows b*T+0 -> lda = T*E)
    gemm_bf16_kernel<false, false><<<dim3(FDd / 128, 1), 256, 0, stream>>>(
        x_bf, Tt * Ee, wt[g * 5 + 2], gin(g, 5), nullptr, 1,
        fproj[g], nullptr, Bb, FDd, Ee);
    // g1 = x @ V + Vb  (bf16)
    gemm_bf16_kernel<true, false><<<dim3(FDd / 128, Mrows / 128), 256, 0, stream>>>(
        x_bf, Ee, wt[g * 5 + 0], gin(g, 1), nullptr, 1,
        nullptr, g1, Mrows, FDd, Ee);
    // g2 = g1 @ S + Sb + fproj[b]  (bf16)
    gemm_bf16_kernel<true, false><<<dim3(FDd / 128, Mrows / 128), 256, 0, stream>>>(
        g1, FDd, wt[g * 5 + 1], gin(g, 3), fproj[g], Tt,
        nullptr, g2, Mrows, FDd, FDd);
    // a_g = g2 @ U + Ub  (fp32)
    gemm_bf16_kernel<false, false><<<dim3(Hh / 128, Mrows / 128), 256, 0, stream>>>(
        g2, FDd, wt[g * 5 + 3], gin(g, 7), nullptr, 1,
        a_g[g], nullptr, Mrows, Hh, FDd);
  }

  // 4) h := h0
  init_h_kernel<<<(Bb * Hh + 255) / 256, 256, 0, stream>>>(h0, hbuf, h_bf);

  // 5) Sequential recurrence: 64 steps x 2 phases (stream order = dependency)
  for (int t = 0; t < Tt; ++t) {
    gru_phase1_kernel<<<16, 256, 0, stream>>>(
        t, hbuf, h_bf, wt[0 * 5 + 4], wt[1 * 5 + 4], gin(0, 9), gin(1, 9),
        a_g[0], a_g[1], z_ws, rh_bf);
    gru_phase2_kernel<<<16, 256, 0, stream>>>(
        t, hbuf, h_bf, wt[2 * 5 + 4], gin(2, 9), a_g[2], z_ws, rh_bf, hs_bf);
  }

  // 6) logits[b,t,:] = hs[t,b,:] @ C_w + C_b   (REMAP handles tb->bt)
  gemm_bf16_kernel<false, true><<<dim3(Vv / 128, Mrows / 128), 256, 0, stream>>>(
      hs_bf, Hh, cw_t, C_b, nullptr, 1, (float*)d_out, nullptr, Mrows, Vv, Hh);
}
